// NetworkOneNeuronGeadah_41180146434430
// MI455X (gfx1250) — matrix operand order repaired and checked by reading, not compiled, not run on hardware
//
#include <hip/hip_runtime.h>
#include <hip/hip_bf16.h>

typedef __attribute__((ext_vector_type(16))) _Float16 v16h;
typedef __attribute__((ext_vector_type(8)))  _Float16 v8h;
typedef __attribute__((ext_vector_type(8)))  float    v8f;

#define T_STEPS 256
#define BATCH   4096
#define HID     64
#define TB      (T_STEPS * BATCH)

// ---- fast math: CDNA5 TRANS32 units directly (no IEEE div expansion) ----
__device__ __forceinline__ float fast_rcp(float x) {
#if __has_builtin(__builtin_amdgcn_rcpf)
    return __builtin_amdgcn_rcpf(x);      // v_rcp_f32
#else
    return 1.0f / x;
#endif
}

#if __has_builtin(__builtin_amdgcn_tanhf)
__device__ __forceinline__ float tanh_fast(float x) {
    return __builtin_amdgcn_tanhf(x);     // v_tanh_f32 (CDNA5 TRANS op)
}
#elif __has_builtin(__builtin_amdgcn_tanh_f32)
__device__ __forceinline__ float tanh_fast(float x) {
    return __builtin_amdgcn_tanh_f32(x);
}
#else
__device__ __forceinline__ float tanh_fast(float x) {
    float e = __expf(2.0f * x);
    return (e - 1.0f) * fast_rcp(e + 1.0f);
}
#endif

__device__ __forceinline__ float sigf(float x) {
    // sigmoid(x) = 0.5 + 0.5*tanh(x/2): 1 trans + 1 fma
    return __builtin_fmaf(0.5f, tanh_fast(0.5f * x), 0.5f);
}

// swap with neighbor lane (xor 1) within 16-lane groups: v_permlane16_b32
__device__ __forceinline__ float perm_xor1(float v) {
#if __has_builtin(__builtin_amdgcn_permlane16)
    unsigned i = __builtin_bit_cast(unsigned, v);
    unsigned p = __builtin_amdgcn_permlane16(i, i, 0x67452301u, 0xEFCDAB89u, false, false);
    return __builtin_bit_cast(float, p);
#else
    return __builtin_bit_cast(float,
        __builtin_amdgcn_ds_swizzle(__builtin_bit_cast(int, v), 0x041f));
#endif
}

__global__ void __launch_bounds__(256)
geadah_kernel(const float* __restrict__ x,
              const float* __restrict__ W_ih,
              const float* __restrict__ W_hh,
              const float* __restrict__ b_ih,
              const float* __restrict__ b_hh,
              const float* __restrict__ W_lin,
              const float* __restrict__ b_lin,
              const float* __restrict__ W_g,
              const float* __restrict__ b_g,
              float* __restrict__ out)
{
    // [buf][rowgroup][m][k] : h state, f16, double-buffered across timesteps
    __shared__ __align__(16) _Float16 h_lds[2][2][16][HID];
    __shared__ float  po_lds[2][16];       // prev_out per row-group
    __shared__ float2 nsp2[2][16];         // (n_t, s_t) per batch row (gw0 scratch)

    const int tid  = threadIdx.x;
    const int lane = tid & 31;
    const int wave = tid >> 5;
    const int rg   = wave >> 2;            // row-group 0..1 (16 batch rows each)
    const int gw   = wave & 3;             // hidden-column group 0..3
    const int lm   = lane & 15;
    const int hi8  = (lane >> 4) << 3;     // C/D layout: row offset 0 or 8
    const int kcol = 16 * gw + lm;         // this lane's hidden column
    const bool lo16 = (lane < 16);

    const int bblk = blockIdx.x * 32;
    const int b0   = bblk + rg * 16;

    const float wg  = W_g[0];
    const float bg  = b_g[0];
    const float bl0 = b_lin[0];
    const float bl1 = b_lin[1];
    // C-seed for the ns-WMMA: column 0 -> b_lin[0], column 1 -> b_lin[1]
    const float nsbias = (lm == 0) ? bl0 : ((lm == 1) ? bl1 : 0.0f);

    // zero both h buffers (h0 = 0)
    {
        unsigned* p = (unsigned*)&h_lds[0][0][0][0];
        #pragma unroll
        for (int i = 0; i < 8; ++i)        // 2*2*16*64 halves = 2048 dwords / 256 thr
            p[tid + i * 256] = 0u;
    }
    // t = 0: n=1, s=1 -> out0 = sigmoid(x0*Wg + bg)
    if (tid < 32) {
        int b = bblk + tid;
        float lin0 = x[b] * wg + bg;
        float o0 = sigf(lin0);
        out[b] = o0;
        out[TB + b] = 1.0f;
        out[2 * TB + b] = 1.0f;
        po_lds[tid >> 4][tid & 15] = o0;
    }

    // ---- per-wave resident weights in WMMA B-operand layout ----
    // B-layout (32x16, f16): lanes 0-15 hold K {0..7}|{16..23},
    // lanes 16-31 hold K {8..15}|{24..31}; N = lane%16.  B[k][n] = W[n][k].
    v16h Bt[4][2];                         // W_hh^T, two K=32 chunks per gate
    v16h BIH[4];                           // W_ih^T in K-rows 0,1 (zeros elsewhere)
    float biasq[4];
    #pragma unroll
    for (int q = 0; q < 4; ++q) {
        const int n = 64 * q + kcol;       // gate column 0..255
        const float* wr = W_hh + n * HID;
        biasq[q] = b_ih[n] + b_hh[n];
        #pragma unroll
        for (int c = 0; c < 2; ++c) {
            const int kb = 32 * c + (lo16 ? 0 : 8);
            v16h t;
            #pragma unroll
            for (int j = 0; j < 16; ++j) t[j] = (_Float16)wr[kb + (j < 8 ? j : j + 8)];
            Bt[q][c] = t;
        }
        v16h ti = {};
        ti[0] = lo16 ? (_Float16)W_ih[2 * n]     : (_Float16)0.0f;   // K=0: x weight
        ti[1] = lo16 ? (_Float16)W_ih[2 * n + 1] : (_Float16)0.0f;   // K=1: prev_out
        BIH[q] = ti;
    }
    // W_lin as a B-operand tile: column 0 = W_lin[0][k], column 1 = W_lin[1][k]
    v16h BW[2];
    {
        const float* wlp = W_lin + ((lm == 1) ? HID : 0);
        const bool act = (lm < 2);
        #pragma unroll
        for (int c = 0; c < 2; ++c) {
            const int kb = 32 * c + (lo16 ? 0 : 8);
            v16h tv;
            #pragma unroll
            for (int j = 0; j < 16; ++j)
                tv[j] = act ? (_Float16)wlp[kb + (j < 8 ? j : j + 8)] : (_Float16)0.0f;
            BW[c] = tv;
        }
    }

    v8f  cst = {};                         // c state tile (16 rows x 16 hidden cols)
    v16h a2  = {};                         // K=2 A chunk: [x_t, prev_out] in K=0,1
    __syncthreads();

    // A-layout (16x32, f16): lanes 0-15 row M=lane, K {0..7}|{16..23};
    // lanes 16-31 row M=lane-16, K {8..15}|{24..31}
    const int arow = lm * HID;
    const int akb  = lo16 ? 0 : 8;

    // ns/out finalizer for time tt from A = h(tt): 2 WMMAs + lane pairing
    auto ns_out = [&](int tt, v16h A0, v16h A1, bool write_po) {
        v8f nsa;
        #pragma unroll
        for (int r = 0; r < 8; ++r) nsa[r] = nsbias;
        nsa = __builtin_amdgcn_wmma_f32_16x16x32_f16(false, A0, false, BW[0],
                                                     (short)0, nsa, false, false);
        nsa = __builtin_amdgcn_wmma_f32_16x16x32_f16(false, A1, false, BW[1],
                                                     (short)0, nsa, false, false);
        // lanes 0/16 hold n_t (col 0), lanes 1/17 hold s_t (col 1)
        #pragma unroll
        for (int r = 0; r < 8; ++r) {
            float nv = nsa[r];
            float sv = perm_xor1(nv);      // lane0 <- lane1's s_t
            if (lm == 0) nsp2[rg][r + hi8] = make_float2(nv, sv);
        }
        float2 ns = nsp2[rg][lm];          // same-wave LDS bounce, 16-wide
        float lin = x[tt * BATCH + b0 + lm] * wg + bg;
        float z  = ns.x * lin;
        float g1 = __logf(1.0f + __expf(z)) * fast_rcp(ns.x);
        float g2 = sigf(z);
        float o  = (1.0f - ns.y) * g1 + ns.y * g2;
        if (lane < 16) {
            int b = b0 + lm;
            out[tt * BATCH + b]          = o;
            out[TB + tt * BATCH + b]     = ns.x;
            out[2 * TB + tt * BATCH + b] = ns.y;
            if (write_po) po_lds[rg][lm] = o;
        }
    };

    for (int t = 1; t < T_STEPS; ++t) {
        const int rbuf = (t - 1) & 1;
        const int wbuf = t & 1;

        // prefetch next step's x slice (speculative; dropped if OOB at t=255)
        __builtin_prefetch((const void*)(x + (t + 1) * BATCH + b0), 0, 1);

        // A operand: h(t-1) (16x64 f16) as two K=32 chunks in ISA A layout
        const _Float16* hb = &h_lds[rbuf][rg][0][0];
        v8h a0l = *(const v8h*)(hb + arow + akb);
        v8h a0h = *(const v8h*)(hb + arow + 16 + akb);
        v8h a1l = *(const v8h*)(hb + arow + 32 + akb);
        v8h a1h = *(const v8h*)(hb + arow + 48 + akb);
        v16h a0 = __builtin_shufflevector(a0l, a0h, 0,1,2,3,4,5,6,7,8,9,10,11,12,13,14,15);
        v16h a1 = __builtin_shufflevector(a1l, a1h, 0,1,2,3,4,5,6,7,8,9,10,11,12,13,14,15);

        // out[t-1] = Geadah(h(t-1) @ W_lin.T + b_lin, x[t-1]) -- one wave per rg
        if (gw == 0 && t > 1) ns_out(t - 1, a0, a1, true);
        __syncthreads();                   // prev_out(t-1) ready for all waves

        // refresh the K=2 A chunk: [x_t, prev_out] (zeros elsewhere, inc. hi lanes)
        {
            float xv  = x[t * BATCH + b0 + lm];
            float pov = po_lds[rg][lm];
            a2[0] = lo16 ? (_Float16)xv  : (_Float16)0.0f;
            a2[1] = lo16 ? (_Float16)pov : (_Float16)0.0f;
        }

        // gates = (b_ih + b_hh) + [x,po] @ W_ih^T + h @ W_hh^T  (all on WMMA)
        v8f acc[4];
        #pragma unroll
        for (int q = 0; q < 4; ++q) {
            v8f c;
            #pragma unroll
            for (int r = 0; r < 8; ++r) c[r] = biasq[q];
            c = __builtin_amdgcn_wmma_f32_16x16x32_f16(false, a2, false, BIH[q],
                                                       (short)0, c, false, false);
            c = __builtin_amdgcn_wmma_f32_16x16x32_f16(false, a0, false, Bt[q][0],
                                                       (short)0, c, false, false);
            c = __builtin_amdgcn_wmma_f32_16x16x32_f16(false, a1, false, Bt[q][1],
                                                       (short)0, c, false, false);
            acc[q] = c;
        }

        // LSTM cell update + h_new(t) to LDS (A operand of step t+1)
        #pragma unroll
        for (int r = 0; r < 8; ++r) {
            float ig = sigf(acc[0][r]);
            float fg = sigf(acc[1][r]);
            float gg = tanh_fast(acc[2][r]);
            float og = sigf(acc[3][r]);
            float c  = fg * cst[r] + ig * gg;
            cst[r]   = c;
            float hn = og * tanh_fast(c);
            h_lds[wbuf][rg][r + hi8][kcol] = (_Float16)hn;
        }
        __syncthreads();                   // h(t) ready for step t+1
    }

    // epilogue: out[T-1] from h(T-1)
    if (gw == 0) {
        const _Float16* hb = &h_lds[(T_STEPS - 1) & 1][rg][0][0];
        v8h a0l = *(const v8h*)(hb + arow + akb);
        v8h a0h = *(const v8h*)(hb + arow + 16 + akb);
        v8h a1l = *(const v8h*)(hb + arow + 32 + akb);
        v8h a1h = *(const v8h*)(hb + arow + 48 + akb);
        v16h a0 = __builtin_shufflevector(a0l, a0h, 0,1,2,3,4,5,6,7,8,9,10,11,12,13,14,15);
        v16h a1 = __builtin_shufflevector(a1l, a1h, 0,1,2,3,4,5,6,7,8,9,10,11,12,13,14,15);
        ns_out(T_STEPS - 1, a0, a1, false);
    }
}

extern "C" void kernel_launch(void* const* d_in, const int* in_sizes, int n_in,
                              void* d_out, int out_size, void* d_ws, size_t ws_size,
                              hipStream_t stream) {
    const float* x     = (const float*)d_in[0];
    const float* W_ih  = (const float*)d_in[1];
    const float* W_hh  = (const float*)d_in[2];
    const float* b_ih  = (const float*)d_in[3];
    const float* b_hh  = (const float*)d_in[4];
    const float* W_lin = (const float*)d_in[5];
    const float* b_lin = (const float*)d_in[6];
    const float* W_g   = (const float*)d_in[7];
    const float* b_g   = (const float*)d_in[8];
    float* out = (float*)d_out;

    dim3 grid(BATCH / 32);                 // 128 blocks x 256 threads (8 wave32)
    dim3 block(256);
    hipLaunchKernelGGL(geadah_kernel, grid, block, 0, stream,
                       x, W_ih, W_hh, b_ih, b_hh, W_lin, b_lin, W_g, b_g, out);
}
